// MultiGranularityRandomGraphAttention_90606630076474
// MI455X (gfx1250) — matrix operand order repaired and checked by reading, not compiled, not run on hardware
//
#include <hip/hip_runtime.h>

// ---------------------------------------------------------------------------
// MultiGranularityRandomGraphAttention for MI455X (gfx1250, wave32, WMMA)
// All matmuls on v_wmma_f32_16x16x32_bf16 (bf16 in, fp32 accumulate).
// v3: WGP-scope prefetch (locality=3) so the next K tile lands in the near
// cache; otherwise identical to v2 (pre-transposed weights, transposed V,
// all inner-loop LDS staging as aligned b128 copies).
// ---------------------------------------------------------------------------

#define BB   32
#define TT   336           // 21 * 16 exactly
#define DD   512
#define HH   8
#define HD   64
#define TPAD 352           // 22*16 = 11*32, padded key dim for P@V chunks
#define NEG_INF (-1e30f)

typedef __attribute__((ext_vector_type(16))) __bf16 v16bf;
typedef __attribute__((ext_vector_type(8)))  __bf16 v8bf;
typedef __attribute__((ext_vector_type(8)))  float  v8f;

union V16U { v16bf v; v8bf h[2]; };

__device__ __forceinline__ __bf16 f2bf(float f) {
    unsigned int u = __builtin_bit_cast(unsigned int, f);
    unsigned int r = (u + 0x7FFFu + ((u >> 16) & 1u)) >> 16;   // RNE
    unsigned short s = (unsigned short)r;
    return __builtin_bit_cast(__bf16, s);
}

__device__ __forceinline__ v8f zero8() {
    v8f z = {0.f, 0.f, 0.f, 0.f, 0.f, 0.f, 0.f, 0.f};
    return z;
}

// A-fragment (16xK tile, row-major, leading dim ld in elements).
// ISA layout: lanes 0-15 -> M=lane, K = {0..7, 16..23}; lanes 16-31 ->
// M=lane-16, K = {8..15, 24..31}.  Two ds_read_b128 per lane.
__device__ __forceinline__ v16bf load_a_frag(const __bf16* tile, int ld) {
    int l = threadIdx.x & 31;
    int row = l & 15;
    int koff = (l < 16) ? 0 : 8;
    const __bf16* p = tile + row * ld + koff;
    V16U u;
    u.h[0] = *(const v8bf*)(p);
    u.h[1] = *(const v8bf*)(p + 16);
    return u.v;
}

// B-fragment (Kx16 tile).  bt is stored column-major: bt[n*ld + k].
// ISA layout: lanes 0-15 -> N=lane, K = {0..7, 16..23}; lanes 16-31 ->
// N=lane-16, K = {8..15, 24..31}.  Two ds_read_b128 per lane.
__device__ __forceinline__ v16bf load_b_frag(const __bf16* bt, int ld) {
    int l = threadIdx.x & 31;
    int n = l & 15;
    int koff = (l < 16) ? 0 : 8;
    const __bf16* p = bt + n * ld + koff;
    V16U u;
    u.h[0] = *(const v8bf*)(p);
    u.h[1] = *(const v8bf*)(p + 16);
    return u.v;
}

// ---------------------------------------------------------------------------
// Kernel 1a: fp32 -> bf16 conversion (elementwise)
// ---------------------------------------------------------------------------
__global__ void cvt_f32_bf16(const float* __restrict__ s,
                             __bf16* __restrict__ d, int n) {
    int i = blockIdx.x * blockDim.x + threadIdx.x;
    int stride = gridDim.x * blockDim.x;
    for (; i < n; i += stride) d[i] = f2bf(s[i]);
}

// ---------------------------------------------------------------------------
// Kernel 1b: fp32 -> bf16 conversion with transpose.
// src is [K][N] row-major fp32; dst is [N][K] row-major bf16.
// block (32,8), grid (N/32, K/32).  LDS-tiled so both sides are coalesced.
// ---------------------------------------------------------------------------
__global__ __launch_bounds__(256)
void cvt_tr_f32_bf16(const float* __restrict__ s,
                     __bf16* __restrict__ d, int K, int N) {
    __shared__ float tile[32][33];
    int n0 = blockIdx.x * 32, k0 = blockIdx.y * 32;
    for (int i = threadIdx.y; i < 32; i += 8)
        tile[i][threadIdx.x] = s[(size_t)(k0 + i) * N + n0 + threadIdx.x];
    __syncthreads();
    for (int i = threadIdx.y; i < 32; i += 8)
        d[(size_t)(n0 + i) * K + k0 + threadIdx.x] = f2bf(tile[threadIdx.x][i]);
}

// ---------------------------------------------------------------------------
// Kernel 2: fused QKV GEMM.  C[m,n] = X[m,:] @ W[:,n], M=B*T, N=K=512.
// Weights are pre-transposed (Wt[n][k]), so A and B staging are identical
// contiguous b128 copies.  grid = (M/64, N/64, 3), block = 128 (4 waves).
// Q,K written bf16 [b,h,t,hd];  V written transposed [b,h,hd,t].
// ---------------------------------------------------------------------------
__global__ __launch_bounds__(128)
void qkv_gemm(const __bf16* __restrict__ xbf,
              const __bf16* __restrict__ wqt,
              const __bf16* __restrict__ wkt,
              const __bf16* __restrict__ wvt,
              __bf16* __restrict__ qb,
              __bf16* __restrict__ kb,
              __bf16* __restrict__ vtb) {
    __shared__ __bf16 ldsA[64 * 32];   // [row][k]
    __shared__ __bf16 ldsB[64 * 32];   // [n][k]

    const int tid = threadIdx.x;
    const int w   = tid >> 5;
    const int l   = tid & 31;
    const int m0  = blockIdx.x * 64;
    const int n0  = blockIdx.y * 64;
    const int z   = blockIdx.z;

    const __bf16* Wt = (z == 0) ? wqt : ((z == 1) ? wkt : wvt);

    const int r  = tid >> 1;            // 0..63
    const int hc = (tid & 1) * 16;      // 0 or 16

    v8f acc[4] = {zero8(), zero8(), zero8(), zero8()};

    for (int kbx = 0; kbx < DD; kbx += 32) {
        {
            const __bf16* srcA = xbf + (size_t)(m0 + r) * DD + kbx + hc;
            v8bf* dstA = (v8bf*)&ldsA[r * 32 + hc];
            dstA[0] = *(const v8bf*)(srcA);
            dstA[1] = *(const v8bf*)(srcA + 8);
            const __bf16* srcB = Wt + (size_t)(n0 + r) * DD + kbx + hc;
            v8bf* dstB = (v8bf*)&ldsB[r * 32 + hc];
            dstB[0] = *(const v8bf*)(srcB);
            dstB[1] = *(const v8bf*)(srcB + 8);
        }
        if (kbx + 32 < DD) {            // WGP-scope prefetch of next K tile
            __builtin_prefetch(xbf + (size_t)(m0 + r) * DD + kbx + 32 + hc, 0, 3);
            __builtin_prefetch(Wt  + (size_t)(n0 + r) * DD + kbx + 32 + hc, 0, 3);
        }
        __syncthreads();

        v16bf a = load_a_frag(&ldsA[(w * 16) * 32], 32);
#pragma unroll
        for (int nt = 0; nt < 4; ++nt) {
            v16bf b = load_b_frag(&ldsB[(nt * 16) * 32], 32);
            acc[nt] = __builtin_amdgcn_wmma_f32_16x16x32_bf16(
                false, a, false, b, (short)0, acc[nt], false, false);
        }
        __syncthreads();
    }

#pragma unroll
    for (int nt = 0; nt < 4; ++nt) {
#pragma unroll
        for (int e = 0; e < 8; ++e) {
            int m   = m0 + w * 16 + e + ((l >= 16) ? 8 : 0);
            int col = n0 + nt * 16 + (l & 15);
            int b   = m / TT, t = m % TT;
            int h   = col >> 6, hd = col & 63;
            __bf16 val = f2bf(acc[nt][e]);
            if (z == 2)   // V transposed: [b,h,hd,t]
                vtb[(((size_t)b * HH + h) * HD + hd) * TT + t] = val;
            else if (z == 1)
                kb[(((size_t)b * HH + h) * TT + t) * HD + hd] = val;
            else
                qb[(((size_t)b * HH + h) * TT + t) * HD + hd] = val;
        }
    }
}

// ---------------------------------------------------------------------------
// Kernel 3: masked-softmax attention for one (b*h, query tile, granularity).
// grid = (B*H, 21, 2), block = 128 (4 waves).
// ---------------------------------------------------------------------------
__global__ __launch_bounds__(128)
void graph_attn(const __bf16* __restrict__ Q,
                const __bf16* __restrict__ K,
                const __bf16* __restrict__ Vt,
                const float*  __restrict__ rbias,
                __bf16* __restrict__ combined) {
    __shared__ __bf16 ldsQ[16 * 64];
    __shared__ __bf16 ldsK[64 * 64];      // 4 key tiles x 64 hd, row-major
    __shared__ float  ldsS[16 * TPAD];
    __shared__ __bf16 ldsP[16 * TPAD];
    __shared__ __bf16 ldsVt[64 * 32];     // [hd][key] for one 32-key chunk
    __shared__ float  red[16 * 8];
    __shared__ float  rowstat[16];

    const int tid = threadIdx.x;
    const int w   = tid >> 5;
    const int l   = tid & 31;
    const int bh  = blockIdx.x;
    const int b   = bh / HH;
    const int h   = bh % HH;
    const int qt  = blockIdx.y;
    const int g   = blockIdx.z;
    const int q0  = qt * 16;
    const __bf16 ZB = __builtin_bit_cast(__bf16, (unsigned short)0);

    // stage Q tile (16x64); init scores to NEG_INF (pad tile 21 stays masked)
    {
        int i = tid * 8;                    // 1024 elems / 128 threads
        int r = i >> 6, c = i & 63;
        const __bf16* src = Q + ((size_t)bh * TT + q0 + r) * HD + c;
        *(v8bf*)&ldsQ[i] = *(const v8bf*)(src);
    }
    for (int i = tid; i < 16 * TPAD; i += 128) ldsS[i] = NEG_INF;
    __syncthreads();

    v16bf a0 = load_a_frag(&ldsQ[0], 64);     // hd 0..31
    v16bf a1 = load_a_frag(&ldsQ[32], 64);    // hd 32..63

    // ---- scores: S = Q K^T / 8 + bias, masked -----------------------------
    for (int ktb = 0; ktb < 21; ktb += 4) {
        // stage 4 key tiles (64 keys x 64 hd), zero-fill past T
        {
            int kk = tid >> 1;
            int hcc = (tid & 1) * 32;
            int key = ktb * 16 + kk;
            v8bf* dst = (v8bf*)&ldsK[kk * 64 + hcc];
            if (key < TT) {
                const __bf16* src = K + ((size_t)bh * TT + key) * HD + hcc;
                dst[0] = *(const v8bf*)(src);
                dst[1] = *(const v8bf*)(src + 8);
                dst[2] = *(const v8bf*)(src + 16);
                dst[3] = *(const v8bf*)(src + 24);
            } else {
                v8bf zv = {ZB, ZB, ZB, ZB, ZB, ZB, ZB, ZB};
                dst[0] = zv; dst[1] = zv; dst[2] = zv; dst[3] = zv;
            }
        }
        __syncthreads();

        int kt = ktb + w;
        if (kt < 21) {
            v16bf b0 = load_b_frag(&ldsK[(w * 16) * 64 + 0], 64);
            v16bf b1 = load_b_frag(&ldsK[(w * 16) * 64 + 32], 64);
            v8f s8 = zero8();
            s8 = __builtin_amdgcn_wmma_f32_16x16x32_bf16(
                false, a0, false, b0, (short)0, s8, false, false);
            s8 = __builtin_amdgcn_wmma_f32_16x16x32_bf16(
                false, a1, false, b1, (short)0, s8, false, false);
#pragma unroll
            for (int e = 0; e < 8; ++e) {
                int rr  = e + ((l >= 16) ? 8 : 0);
                int qr  = q0 + rr;
                int key = kt * 16 + (l & 15);
                float sc = s8[e] * 0.125f
                         + rbias[((size_t)g * TT + qr) * TT + key];
                bool adj = (g == 0) ? ((qr % 24) == (key % 24))
                                    : (((qr / 24) % 7) == ((key / 24) % 7));
                ldsS[rr * TPAD + key] = adj ? sc : NEG_INF;
            }
        }
        __syncthreads();
    }

    // ---- softmax over 352 (pad -> exp 0) ----------------------------------
    {
        int row = tid >> 3, sub = tid & 7, c0 = sub * 44;
        float lm = NEG_INF;
        for (int c = c0; c < c0 + 44; ++c) lm = fmaxf(lm, ldsS[row * TPAD + c]);
        red[row * 8 + sub] = lm;
        __syncthreads();
        if (sub == 0) {
            float m = red[row * 8];
            for (int j = 1; j < 8; ++j) m = fmaxf(m, red[row * 8 + j]);
            rowstat[row] = m;
        }
        __syncthreads();
        float rm = rowstat[row];
        float ls = 0.f;
        for (int c = c0; c < c0 + 44; ++c) {
            float e = __expf(ldsS[row * TPAD + c] - rm);
            ldsS[row * TPAD + c] = e;
            ls += e;
        }
        red[row * 8 + sub] = ls;
        __syncthreads();
        if (sub == 0) {
            float s = 0.f;
            for (int j = 0; j < 8; ++j) s += red[row * 8 + j];
            rowstat[row] = s;
        }
        __syncthreads();
        float inv = 1.0f / rowstat[row];
        for (int c = c0; c < c0 + 44; ++c)
            ldsP[row * TPAD + c] = f2bf(ldsS[row * TPAD + c] * inv);
    }
    __syncthreads();

    // ---- O = P @ V : 11 chunks of K=32 keys; wave w owns hd slice w*16 ----
    v8f ao = zero8();
    for (int ch = 0; ch < 11; ++ch) {
        int key0 = ch * 32;
        // stage V chunk from transposed global layout: plain b128 row copies
        {
            int hd = tid >> 1;
            int kh = (tid & 1) * 16;
            const __bf16* src = Vt + ((size_t)bh * HD + hd) * TT + key0 + kh;
            __bf16* dst = &ldsVt[hd * 32 + kh];
            if (key0 + kh + 16 <= TT) {
                *(v8bf*)(dst)     = *(const v8bf*)(src);
                *(v8bf*)(dst + 8) = *(const v8bf*)(src + 8);
            } else {
#pragma unroll
                for (int j = 0; j < 16; ++j) {
                    int key = key0 + kh + j;
                    dst[j] = (key < TT) ? src[j] : ZB;
                }
            }
        }
        __syncthreads();

        v16bf ap = load_a_frag(&ldsP[key0], TPAD);
        v16bf bv = load_b_frag(&ldsVt[(w * 16) * 32], 32);
        ao = __builtin_amdgcn_wmma_f32_16x16x32_bf16(
            false, ap, false, bv, (short)0, ao, false, false);
        __syncthreads();
    }

    // write bf16 into concat buffer [B, T, 2D]: hour at [0,D), day at [D,2D)
#pragma unroll
    for (int e = 0; e < 8; ++e) {
        int qr = q0 + e + ((l >= 16) ? 8 : 0);
        int hd = w * 16 + (l & 15);
        combined[((size_t)b * TT + qr) * (2 * DD) + (size_t)g * DD + h * HD + hd]
            = f2bf(ao[e]);
    }
}

// ---------------------------------------------------------------------------
// Kernel 4: output projection.  out = combined(bf16)[M,1024] @ Wo + bo.
// Wo pre-transposed: Wot[n][k], k in [0,1024).  grid = (M/64, 512/64).
// ---------------------------------------------------------------------------
__global__ __launch_bounds__(128)
void out_gemm(const __bf16* __restrict__ cmb,
              const __bf16* __restrict__ wot,
              const float*  __restrict__ bo,
              float* __restrict__ out) {
    __shared__ __bf16 ldsA[64 * 32];
    __shared__ __bf16 ldsB[64 * 32];

    const int tid = threadIdx.x;
    const int w   = tid >> 5;
    const int l   = tid & 31;
    const int m0  = blockIdx.x * 64;
    const int n0  = blockIdx.y * 64;

    const int r  = tid >> 1;
    const int hc = (tid & 1) * 16;

    v8f acc[4] = {zero8(), zero8(), zero8(), zero8()};

    for (int kbx = 0; kbx < 2 * DD; kbx += 32) {
        {
            const __bf16* srcA = cmb + (size_t)(m0 + r) * (2 * DD) + kbx + hc;
            v8bf* dstA = (v8bf*)&ldsA[r * 32 + hc];
            dstA[0] = *(const v8bf*)(srcA);
            dstA[1] = *(const v8bf*)(srcA + 8);
            const __bf16* srcB = wot + (size_t)(n0 + r) * (2 * DD) + kbx + hc;
            v8bf* dstB = (v8bf*)&ldsB[r * 32 + hc];
            dstB[0] = *(const v8bf*)(srcB);
            dstB[1] = *(const v8bf*)(srcB + 8);
        }
        if (kbx + 32 < 2 * DD) {        // WGP-scope prefetch of next K tile
            __builtin_prefetch(cmb + (size_t)(m0 + r) * (2 * DD) + kbx + 32 + hc, 0, 3);
            __builtin_prefetch(wot + (size_t)(n0 + r) * (2 * DD) + kbx + 32 + hc, 0, 3);
        }
        __syncthreads();

        v16bf a = load_a_frag(&ldsA[(w * 16) * 32], 32);
#pragma unroll
        for (int nt = 0; nt < 4; ++nt) {
            v16bf b = load_b_frag(&ldsB[(nt * 16) * 32], 32);
            acc[nt] = __builtin_amdgcn_wmma_f32_16x16x32_bf16(
                false, a, false, b, (short)0, acc[nt], false, false);
        }
        __syncthreads();
    }

#pragma unroll
    for (int nt = 0; nt < 4; ++nt) {
#pragma unroll
        for (int e = 0; e < 8; ++e) {
            int m   = m0 + w * 16 + e + ((l >= 16) ? 8 : 0);
            int col = n0 + nt * 16 + (l & 15);
            out[(size_t)m * DD + col] = acc[nt][e] + bo[col];
        }
    }
}

// ---------------------------------------------------------------------------
// Launch
// ---------------------------------------------------------------------------
extern "C" void kernel_launch(void* const* d_in, const int* in_sizes, int n_in,
                              void* d_out, int out_size, void* d_ws, size_t ws_size,
                              hipStream_t stream) {
    const float* x  = (const float*)d_in[0];
    const float* Wq = (const float*)d_in[1];
    const float* Wk = (const float*)d_in[2];
    const float* Wv = (const float*)d_in[3];
    const float* Wo = (const float*)d_in[4];
    const float* bo = (const float*)d_in[5];
    const float* rb = (const float*)d_in[6];
    float* out = (float*)d_out;

    const size_t NX  = (size_t)BB * TT * DD;     // 5,505,024
    const size_t NW  = (size_t)DD * DD;          // 262,144
    const size_t NWO = (size_t)2 * DD * DD;      // 524,288
    const size_t NC  = (size_t)BB * TT * 2 * DD; // 11,010,048

    size_t off = 0;
    auto carve = [&](size_t bytes) -> void* {
        void* p = (char*)d_ws + off;
        off += (bytes + 255) & ~(size_t)255;
        return p;
    };
    __bf16* xbf = (__bf16*)carve(NX * 2);
    __bf16* wqt = (__bf16*)carve(NW * 2);
    __bf16* wkt = (__bf16*)carve(NW * 2);
    __bf16* wvt = (__bf16*)carve(NW * 2);
    __bf16* wot = (__bf16*)carve(NWO * 2);
    __bf16* qb  = (__bf16*)carve(NX * 2);
    __bf16* kb  = (__bf16*)carve(NX * 2);
    __bf16* vtb = (__bf16*)carve(NX * 2);
    __bf16* cmb = (__bf16*)carve(NC * 2);
    (void)ws_size;

    cvt_f32_bf16<<<1024, 256, 0, stream>>>(x, xbf, (int)NX);

    dim3 cb(32, 8);
    cvt_tr_f32_bf16<<<dim3(16, 16), cb, 0, stream>>>(Wq, wqt, DD, DD);
    cvt_tr_f32_bf16<<<dim3(16, 16), cb, 0, stream>>>(Wk, wkt, DD, DD);
    cvt_tr_f32_bf16<<<dim3(16, 16), cb, 0, stream>>>(Wv, wvt, DD, DD);
    cvt_tr_f32_bf16<<<dim3(16, 32), cb, 0, stream>>>(Wo, wot, 2 * DD, DD);

    dim3 gQKV((BB * TT) / 64, DD / 64, 3);       // (168, 8, 3)
    qkv_gemm<<<gQKV, 128, 0, stream>>>(xbf, wqt, wkt, wvt, qb, kb, vtb);

    dim3 gAtt(BB * HH, TT / 16, 2);              // (256, 21, 2)
    graph_attn<<<gAtt, 128, 0, stream>>>(qb, kb, vtb, rb, cmb);

    dim3 gOut((BB * TT) / 64, DD / 64, 1);       // (168, 8)
    out_gemm<<<gOut, 128, 0, stream>>>(cmb, wot, bo, out);

    (void)in_sizes; (void)n_in; (void)out_size;
}
